// CausalBankModel_34325378629767
// MI455X (gfx1250) — matrix operand order repaired and checked by reading, not compiled, not run on hardware
//
#include <hip/hip_runtime.h>

// ---------------------------------------------------------------------------
// CausalBankModel for MI455X (gfx1250): bf16 WMMA GEMMs + chunked linear scan.
// - v_wmma_f32_16x16x32_bf16 for all GEMMs (fp32 accumulate)
// - double-buffered LDS staging; A panels via GLOBAL_LOAD_ASYNC_TO_LDS_B128
// - fp32->bf16 conversion via native __bf16 cast (backend picks HW cvt)
// ---------------------------------------------------------------------------

typedef __attribute__((ext_vector_type(16))) __bf16          v16bf;
typedef __attribute__((ext_vector_type(16))) unsigned short  v16u;
typedef __attribute__((ext_vector_type(8)))  float           v8f;
typedef int v4i_ __attribute__((vector_size(16)));   // matches async builtin param

#if __has_builtin(__builtin_amdgcn_global_load_async_to_lds_b128) && \
    __has_builtin(__builtin_amdgcn_s_wait_asynccnt)
#define HAS_ASYNC_LDS 1
#else
#define HAS_ASYNC_LDS 0
#endif

constexpr int BB   = 2;
constexpr int SS   = 1024;
constexpr int DD   = 512;
constexpr int MM   = 1024;
constexpr int HH   = 1024;
constexpr int EE   = 4;
constexpr int VV   = 32000;
constexpr int NROW = BB * SS;      // 2048 tokens
constexpr int FF   = MM + DD;      // 1536 feature dim
constexpr int KALL = EE * HH;      // 4096 concatenated expert hidden
constexpr int NCH  = 16;           // scan chunks
constexpr int CHT  = SS / NCH;     // 64 steps per chunk

// fp32 -> bf16 via native cast: lets the backend select hardware cvt ops
static __device__ __forceinline__ unsigned short f2bf(float f) {
  return __builtin_bit_cast(unsigned short, (__bf16)f);
}
static __device__ __forceinline__ unsigned int f2bf_pk(float lo, float hi) {
  return (unsigned int)f2bf(lo) | ((unsigned int)f2bf(hi) << 16);
}
static __device__ __forceinline__ float bf2f(unsigned short h) {
  unsigned int u = ((unsigned int)h) << 16;
  return __builtin_bit_cast(float, u);
}

// ---------------------------------------------------------------------------
// 1) Embedding gather -> bf16 into featB[:, M:M+D]  (packed pair stores)
// ---------------------------------------------------------------------------
__global__ void gather_embed_k(const int* __restrict__ tokens,
                               const float* __restrict__ embed,
                               unsigned short* __restrict__ featB) {
  const int n   = blockIdx.x;
  const int tok = tokens[n];
  const float* src = embed + (size_t)tok * DD;
  unsigned short* dst = featB + (size_t)n * FF + MM;
  const int d = threadIdx.x * 2;             // 256 threads x 2 elems = 512
  if (d < DD) {
    const float2 v = *(const float2*)&src[d];
    *(unsigned int*)&dst[d] = f2bf_pk(v.x, v.y);
  }
}

// ---------------------------------------------------------------------------
// 2/5/6) Tiled WMMA GEMM: C[128x128] per workgroup, K-step 32, bf16 inputs.
//   A: bf16 (ushort) in memory, row-major (lda) -> async copy into LDS.
//   B: fp32 in memory, row-major [K x N] (ldb) -> bf16 while staged (transposed).
// MODE 0: out fp32 = A@B                      (u projection)
// MODE 1: out fp32 = sigmoid(A@B + bias[col]) (decay gate a)
// MODE 2: outb bf16 = gate * relu(A@B_e + b1_e[col])^2, expert e = blockIdx.z
// MODE 3: out fp32 = A@B + sum_e gates[row,e]*b2[e,col]   (logits)
// ---------------------------------------------------------------------------
template <int MODE>
__global__ __launch_bounds__(256) void gemm_wmma_k(
    const unsigned short* __restrict__ A, int lda, int K,
    const float* __restrict__ Bw, int ldb,
    float* __restrict__ out, int ldo,
    const float* __restrict__ bias,
    const float* __restrict__ gates,
    const float* __restrict__ b2,
    unsigned short* __restrict__ outb) {
  __shared__ unsigned short As[2][128][48];   // [buf][row][k]  (96B row stride)
  __shared__ unsigned short Bt[2][128][48];   // [buf][col][k]  (K-transposed)

  const int colBase = blockIdx.x * 128;
  const int rowBase = blockIdx.y * 128;
  const int expert  = blockIdx.z;
  if (MODE == 2) Bw += (size_t)expert * FF * HH;   // w1[e]

  const int tid  = threadIdx.x;
  const int lane = tid & 31;
  const int w    = tid >> 5;          // 8 waves
  const int wr   = w >> 2;            // 0..1 : rows 64*wr
  const int wc   = w & 3;             // 0..3 : cols 32*wc

  // staging responsibilities
  const int r_a  = tid >> 1;          // A: 2 threads per row, 16 ushorts each
  const int cb_a = (tid & 1) * 16;
  const int kr_b = tid >> 3;          // B: 8 threads per k-row, 16 floats each
  const int c0_b = (tid & 7) * 16;

  v8f acc[4][2];
#pragma unroll
  for (int i = 0; i < 4; ++i)
#pragma unroll
    for (int j = 0; j < 2; ++j) acc[i][j] = (v8f)0.0f;

  float tb[16];                       // B panel in flight (regs)
#if !HAS_ASYNC_LDS
  uint4 qa0, qa1;                     // A panel in flight (regs, fallback)
#endif

  // ---- issue A-panel copy for K-step k0 into LDS buffer bufi ----
  auto issueA = [&](int bufi, int k0) {
    const unsigned short* ap = A + (size_t)(rowBase + r_a) * lda + k0 + cb_a;
#if HAS_ASYNC_LDS
    __builtin_amdgcn_global_load_async_to_lds_b128(
        (v4i_*)ap, (v4i_*)&As[bufi][r_a][cb_a], 0, 0);
    __builtin_amdgcn_global_load_async_to_lds_b128(
        (v4i_*)(ap + 8), (v4i_*)&As[bufi][r_a][cb_a + 8], 0, 0);
#else
    (void)bufi;
    qa0 = ((const uint4*)ap)[0];
    qa1 = ((const uint4*)ap)[1];
#endif
  };
  // ---- issue B-panel global loads for K-step k0 into registers ----
  auto loadB = [&](int k0) {
    const float* bp = Bw + (size_t)(k0 + kr_b) * ldb + colBase + c0_b;
    *(float4*)&tb[0]  = ((const float4*)bp)[0];
    *(float4*)&tb[4]  = ((const float4*)bp)[1];
    *(float4*)&tb[8]  = ((const float4*)bp)[2];
    *(float4*)&tb[12] = ((const float4*)bp)[3];
    if (k0 + 32 < K)  // hint next-next panel (global_prefetch_b8)
      __builtin_prefetch(bp + (size_t)32 * ldb, 0, 1);
  };
  // ---- write staged panels into LDS buffer bufi ----
  auto writeStage = [&](int bufi) {
#if !HAS_ASYNC_LDS
    *(uint4*)&As[bufi][r_a][cb_a]     = qa0;
    *(uint4*)&As[bufi][r_a][cb_a + 8] = qa1;
#endif
#pragma unroll
    for (int j = 0; j < 16; j += 2) {
      const unsigned int pk = f2bf_pk(tb[j], tb[j + 1]);
      Bt[bufi][c0_b + j][kr_b]     = (unsigned short)(pk & 0xFFFFu);
      Bt[bufi][c0_b + j + 1][kr_b] = (unsigned short)(pk >> 16);
    }
  };
  // ---- 8 WMMAs from LDS buffer bufi ----
  auto compute = [&](int bufi) {
    v16bf bfrag[2];
#pragma unroll
    for (int nt = 0; nt < 2; ++nt) {
      const int bcol = 32 * wc + 16 * nt + (lane & 15);
      const int kbb  = (lane >> 4) * 8;
      v16u bu;
#pragma unroll
      for (int p = 0; p < 8; ++p) {
        unsigned int d = *(const unsigned int*)&Bt[bufi][bcol][((p >= 4) ? 16 : 0) + kbb + (p & 3) * 2];
        bu[2 * p]     = (unsigned short)(d & 0xffffu);
        bu[2 * p + 1] = (unsigned short)(d >> 16);
      }
      bfrag[nt] = __builtin_bit_cast(v16bf, bu);
    }
#pragma unroll
    for (int mt = 0; mt < 4; ++mt) {
      const int arow = 64 * wr + 16 * mt + (lane & 15);
      const int kb   = (lane >> 4) * 8;
      v16u au;
#pragma unroll
      for (int p = 0; p < 8; ++p) {
        const int k = ((p >= 4) ? 16 : 0) + kb + (p & 3) * 2;
        unsigned int d = *(const unsigned int*)&As[bufi][arow][k];
        au[2 * p]     = (unsigned short)(d & 0xffffu);
        au[2 * p + 1] = (unsigned short)(d >> 16);
      }
      v16bf afrag = __builtin_bit_cast(v16bf, au);
      acc[mt][0] = __builtin_amdgcn_wmma_f32_16x16x32_bf16(
          false, afrag, false, bfrag[0], (short)0, acc[mt][0], false, false);
      acc[mt][1] = __builtin_amdgcn_wmma_f32_16x16x32_bf16(
          false, afrag, false, bfrag[1], (short)0, acc[mt][1], false, false);
    }
  };

  // -------- prologue: stage K-step 0 into buffer 0 --------
  issueA(0, 0);
  loadB(0);
  writeStage(0);
#if HAS_ASYNC_LDS
  __builtin_amdgcn_s_wait_asynccnt(0);
#endif
  __syncthreads();

  // -------- pipelined main loop: one barrier per K-step --------
  int buf = 0;
  for (int k0 = 0; k0 < K; k0 += 32) {
    const bool more = (k0 + 32) < K;
    if (more) {                 // start next panel's memory traffic first
      issueA(buf ^ 1, k0 + 32);
      loadB(k0 + 32);
    }
    compute(buf);               // WMMAs overlap the in-flight loads
    if (more) writeStage(buf ^ 1);
#if HAS_ASYNC_LDS
    __builtin_amdgcn_s_wait_asynccnt(0);
#endif
    __syncthreads();
    buf ^= 1;
  }

  // ---- epilogue: C layout -> VGPR r holds M=r (lanes 0-15) / M=r+8 (16-31) --
#pragma unroll
  for (int mt = 0; mt < 4; ++mt) {
#pragma unroll
    for (int nt = 0; nt < 2; ++nt) {
#pragma unroll
      for (int r = 0; r < 8; ++r) {
        const int grow = rowBase + 64 * wr + 16 * mt + r + ((lane >= 16) ? 8 : 0);
        const int gcol = colBase + 32 * wc + 16 * nt + (lane & 15);
        float x = acc[mt][nt][r];
        if (MODE == 0) {
          out[(size_t)grow * ldo + gcol] = x;
        } else if (MODE == 1) {
          x += bias[gcol];
          out[(size_t)grow * ldo + gcol] = 1.0f / (1.0f + __expf(-x));
        } else if (MODE == 2) {
          x += bias[(size_t)expert * HH + gcol];
          x = fmaxf(x, 0.0f);
          x = x * x * gates[(size_t)grow * EE + expert];
          outb[(size_t)grow * KALL + (size_t)expert * HH + gcol] = f2bf(x);
        } else {  // MODE 3
          const float* g = gates + (size_t)grow * EE;
          x += g[0] * b2[gcol] + g[1] * b2[(size_t)VV + gcol] +
               g[2] * b2[2 * (size_t)VV + gcol] + g[3] * b2[3 * (size_t)VV + gcol];
          out[(size_t)grow * ldo + gcol] = x;
        }
      }
    }
  }
}

// ---------------------------------------------------------------------------
// 3) Chunked linear recurrence h_t = a_t*h_{t-1} + u_t  (3-phase scan)
// ---------------------------------------------------------------------------
__global__ void scan_phase1_k(const float* __restrict__ a, const float* __restrict__ u,
                              float* __restrict__ cA, float* __restrict__ cU) {
  const int tid = blockIdx.x * blockDim.x + threadIdx.x;  // b*NCH*M + c*M + m
  const int m = tid & (MM - 1);
  const int c = (tid >> 10) & (NCH - 1);
  const int b = tid >> 14;
  const size_t base = ((size_t)(b * SS + c * CHT)) * MM + m;
  float A = 1.0f, U = 0.0f;
  for (int s = 0; s < CHT; ++s) {
    const float av = a[base + (size_t)s * MM];
    const float uv = u[base + (size_t)s * MM];
    A *= av;
    U = U * av + uv;
  }
  cA[tid] = A;
  cU[tid] = U;
}

__global__ void scan_phase2_k(const float* __restrict__ cA, const float* __restrict__ cU,
                              float* __restrict__ hin) {
  const int tid = blockIdx.x * blockDim.x + threadIdx.x;  // b*M + m
  const int m = tid & (MM - 1);
  const int b = tid >> 10;
  float h = 0.0f;
  for (int c = 0; c < NCH; ++c) {
    const size_t idx = (size_t)(b * NCH + c) * MM + m;
    hin[idx] = h;
    h = cA[idx] * h + cU[idx];
  }
}

__global__ void scan_phase3_k(const float* __restrict__ a, const float* __restrict__ u,
                              const float* __restrict__ hin,
                              unsigned short* __restrict__ featB) {
  const int tid = blockIdx.x * blockDim.x + threadIdx.x;
  const int m = tid & (MM - 1);
  const int c = (tid >> 10) & (NCH - 1);
  const int b = tid >> 14;
  const size_t base = ((size_t)(b * SS + c * CHT)) * MM + m;
  float h = hin[tid];
  for (int s = 0; s < CHT; ++s) {
    const float av = a[base + (size_t)s * MM];
    const float uv = u[base + (size_t)s * MM];
    h = av * h + uv;
    featB[((size_t)(b * SS + c * CHT + s)) * FF + m] = f2bf(h);
  }
}

// ---------------------------------------------------------------------------
// 4) Router: gates = softmax(feat @ router_w + router_b), one wave per token
// ---------------------------------------------------------------------------
__global__ void router_k(const unsigned short* __restrict__ featB,
                         const float* __restrict__ rw, const float* __restrict__ rb,
                         float* __restrict__ gates) {
  const int wave = (blockIdx.x * blockDim.x + threadIdx.x) >> 5;
  const int lane = threadIdx.x & 31;
  if (wave >= NROW) return;
  const unsigned short* f = featB + (size_t)wave * FF;
  float acc[EE] = {0.f, 0.f, 0.f, 0.f};
  for (int j = lane; j < FF; j += 32) {
    const float x = bf2f(f[j]);
    const float* wp = rw + (size_t)j * EE;
    acc[0] += x * wp[0]; acc[1] += x * wp[1];
    acc[2] += x * wp[2]; acc[3] += x * wp[3];
  }
#pragma unroll
  for (int off = 16; off >= 1; off >>= 1)
#pragma unroll
    for (int e = 0; e < EE; ++e) acc[e] += __shfl_down(acc[e], off, 32);
  if (lane == 0) {
    float mx = -1e30f;
#pragma unroll
    for (int e = 0; e < EE; ++e) { acc[e] += rb[e]; mx = fmaxf(mx, acc[e]); }
    float s = 0.f;
#pragma unroll
    for (int e = 0; e < EE; ++e) { acc[e] = __expf(acc[e] - mx); s += acc[e]; }
    const float inv = 1.0f / s;
#pragma unroll
    for (int e = 0; e < EE; ++e) gates[(size_t)wave * EE + e] = acc[e] * inv;
  }
}

// ---------------------------------------------------------------------------
extern "C" void kernel_launch(void* const* d_in, const int* in_sizes, int n_in,
                              void* d_out, int out_size, void* d_ws, size_t ws_size,
                              hipStream_t stream) {
  const int*   tokens   = (const int*)d_in[0];
  const float* embed    = (const float*)d_in[1];
  const float* in_proj  = (const float*)d_in[2];
  const float* gate_w   = (const float*)d_in[3];
  const float* gate_b   = (const float*)d_in[4];
  const float* router_w = (const float*)d_in[5];
  const float* router_b = (const float*)d_in[6];
  const float* w1       = (const float*)d_in[7];
  const float* b1       = (const float*)d_in[8];
  const float* w2       = (const float*)d_in[9];
  const float* b2       = (const float*)d_in[10];
  float* out = (float*)d_out;

  char* ws = (char*)d_ws;
  size_t off = 0;
  auto carve = [&](size_t bytes) -> void* {
    void* p = ws + off;
    off += (bytes + 255) & ~(size_t)255;
    return p;
  };
  unsigned short* featB = (unsigned short*)carve((size_t)NROW * FF * 2);   //  6 MB
  float* u    = (float*)carve((size_t)NROW * MM * 4);                      //  8 MB
  float* a    = (float*)carve((size_t)NROW * MM * 4);                      //  8 MB
  float* cA   = (float*)carve((size_t)BB * NCH * MM * 4);
  float* cU   = (float*)carve((size_t)BB * NCH * MM * 4);
  float* hin  = (float*)carve((size_t)BB * NCH * MM * 4);
  float* gate = (float*)carve((size_t)NROW * EE * 4);
  unsigned short* hidB = (unsigned short*)carve((size_t)NROW * KALL * 2);  // 16 MB

  // 1) gather embeddings (bf16 right half of featB)
  gather_embed_k<<<NROW, 256, 0, stream>>>(tokens, embed, featB);

  // 2) u = emb@in_proj ; a = sigmoid(emb@gate_w + gate_b)
  dim3 gUA(MM / 128, NROW / 128, 1);
  gemm_wmma_k<0><<<gUA, 256, 0, stream>>>(featB + MM, FF, DD, in_proj, MM,
                                          u, MM, nullptr, nullptr, nullptr, nullptr);
  gemm_wmma_k<1><<<gUA, 256, 0, stream>>>(featB + MM, FF, DD, gate_w, MM,
                                          a, MM, gate_b, nullptr, nullptr, nullptr);

  // 3) chunked scan -> h (bf16 left half of featB)
  scan_phase1_k<<<(BB * NCH * MM) / 256, 256, 0, stream>>>(a, u, cA, cU);
  scan_phase2_k<<<(BB * MM) / 256, 256, 0, stream>>>(cA, cU, hin);
  scan_phase3_k<<<(BB * NCH * MM) / 256, 256, 0, stream>>>(a, u, hin, featB);

  // 4) router softmax gates
  router_k<<<NROW / 8, 256, 0, stream>>>(featB, router_w, router_b, gate);

  // 5) per-expert MLP hidden (bf16, experts concatenated along K)
  dim3 gH(HH / 128, NROW / 128, EE);
  gemm_wmma_k<2><<<gH, 256, 0, stream>>>(featB, FF, FF, w1, HH,
                                         nullptr, 0, b1, gate, nullptr, hidB);

  // 6) logits = hid @ w2 + gates @ b2
  dim3 gL(VV / 128, NROW / 128, 1);
  gemm_wmma_k<3><<<gL, 256, 0, stream>>>(hidB, KALL, KALL, w2, VV,
                                         out, VV, nullptr, gate, b2, nullptr);

  (void)in_sizes; (void)n_in; (void)out_size; (void)ws_size;
}
